// Discriminator_11708080849203
// MI455X (gfx1250) — matrix operand order repaired and checked by reading, not compiled, not run on hardware
//
#include <hip/hip_runtime.h>
#include <hip/hip_bf16.h>
#include <math.h>

typedef __attribute__((ext_vector_type(16))) _Float16 v16h;
typedef __attribute__((ext_vector_type(8)))  _Float16 v8h;
typedef __attribute__((ext_vector_type(8)))  float    v8f;

#define N_ROWS 8192
#define DIMD   128
#define INV_TAU 2.0f        // 1/0.5
#define PF_DIST 8           // prefetch distance in 16-row tiles
#define NTILES  (N_ROWS / 16)          // 512 column tiles
#define SPLITS  16                     // j-sweep split factor
#define TILES_PER_CHUNK (NTILES / SPLITS)  // 32

// ---------------------------------------------------------------------------
// Kernel 1: h = fc2(elu(fc1(z))) ; row-L2-normalize ; emit f16 row-major.
// One 128-thread block per row (gridDim.x = 2*8192).
// ---------------------------------------------------------------------------
__global__ __launch_bounds__(128) void proj_norm_kernel(
    const float* __restrict__ z1, const float* __restrict__ z2,
    const float* __restrict__ w1, const float* __restrict__ b1,
    const float* __restrict__ w2, const float* __restrict__ b2,
    _Float16* __restrict__ h1n, _Float16* __restrict__ h2n)
{
    const int gid   = blockIdx.x;
    const int which = gid >> 13;            // 0 -> z1, 1 -> z2
    const int row   = gid & (N_ROWS - 1);
    const float*   z   = which ? z2  : z1;
    _Float16*      out = which ? h2n : h1n;
    const int j = threadIdx.x;

    __shared__ float zs[DIMD];
    __shared__ float hs[DIMD];
    __shared__ float red[DIMD];

    zs[j] = z[(size_t)row * DIMD + j];
    __syncthreads();

    const float* w1r = w1 + (size_t)j * DIMD;
    float acc = b1[j];
    #pragma unroll 8
    for (int k = 0; k < DIMD; ++k) acc = fmaf(zs[k], w1r[k], acc);
    hs[j] = acc > 0.0f ? acc : (__expf(acc) - 1.0f);   // ELU(alpha=1)
    __syncthreads();

    const float* w2r = w2 + (size_t)j * DIMD;
    float o = b2[j];
    #pragma unroll 8
    for (int k = 0; k < DIMD; ++k) o = fmaf(hs[k], w2r[k], o);

    red[j] = o * o;
    __syncthreads();
    for (int s = 64; s > 0; s >>= 1) {
        if (j < s) red[j] += red[j + s];
        __syncthreads();
    }
    const float invn = 1.0f / fmaxf(sqrtf(red[0]), 1e-12f);
    out[(size_t)row * DIMD + j] = (_Float16)(o * invn);
}

// ---------------------------------------------------------------------------
// Kernel 2: partial exp-row-sums. Grid = 64*SPLITS blocks of 8 waves.
// Wave (rowTile, chunk) sweeps 32 column tiles with 16 WMMAs each and writes
// per-row partial sums {S11,S12,S21,S22} to ws. chunk = blockIdx/64 so that
// co-resident blocks stream identical B tiles (WGP$/L2 dedup).
// ---------------------------------------------------------------------------
__global__ __launch_bounds__(256) void semi_loss_partial_kernel(
    const _Float16* __restrict__ h1, const _Float16* __restrict__ h2,
    float4* __restrict__ psums)
{
    const int lane    = threadIdx.x & 31;
    const int wave    = threadIdx.x >> 5;
    const int chunk   = blockIdx.x >> 6;                    // 0..SPLITS-1
    const int rowTile = ((blockIdx.x & 63) << 3) + wave;    // 0..511
    const int row0    = rowTile * 16;
    const int lhalf   = lane >> 4;                          // 0 or 1
    const int lmod    = lane & 15;

    // ---- A tiles (16x32 f16 per k-step): lane holds row (row0+lmod),
    //      halves 0..15 = K kb*32 + lhalf*16 .. +15 (contiguous).
    const _Float16* A1 = h1 + (size_t)(row0 + lmod) * DIMD;
    const _Float16* A2 = h2 + (size_t)(row0 + lmod) * DIMD;
    v16h a1[4], a2[4];
    #pragma unroll
    for (int kb = 0; kb < 4; ++kb) {
        const int lo = kb * 32 + lhalf * 16;
        v8h l1 = *(const v8h*)(A1 + lo);
        v8h u1 = *(const v8h*)(A1 + lo + 8);
        v8h l2 = *(const v8h*)(A2 + lo);
        v8h u2 = *(const v8h*)(A2 + lo + 8);
        a1[kb] = __builtin_shufflevector(l1, u1, 0,1,2,3,4,5,6,7,8,9,10,11,12,13,14,15);
        a2[kb] = __builtin_shufflevector(l2, u2, 0,1,2,3,4,5,6,7,8,9,10,11,12,13,14,15);
    }

    // per-lane partial row sums of exp(S/tau): vgpr r <-> row r + 8*lhalf
    float s11[8], s12[8], s21[8], s22[8];
    #pragma unroll
    for (int r = 0; r < 8; ++r) { s11[r] = 0.f; s12[r] = 0.f; s21[r] = 0.f; s22[r] = 0.f; }

    const size_t laneRowOff = (size_t)lmod * DIMD;
    const int jt0 = chunk * TILES_PER_CHUNK;

    for (int j = 0; j < TILES_PER_CHUNK; ++j) {
        const int jt = jt0 + j;
        const _Float16* B1 = h1 + (size_t)jt * 16 * DIMD + laneRowOff;
        const _Float16* B2 = h2 + (size_t)jt * 16 * DIMD + laneRowOff;

        // prefetch future tiles into WGP$/L2 (global_prefetch_b8, no counter)
        if (jt + PF_DIST < NTILES) {
            __builtin_prefetch(B1 + (size_t)PF_DIST * 16 * DIMD + lhalf * 64, 0, 0);
            __builtin_prefetch(B2 + (size_t)PF_DIST * 16 * DIMD + lhalf * 64, 0, 0);
        }

        // phase 1: B loads
        v16h b1v[4], b2v[4];
        #pragma unroll
        for (int kb = 0; kb < 4; ++kb) {
            const int off = kb * 32 + lhalf * 16;
            b1v[kb] = *(const v16h*)(B1 + off);
            b2v[kb] = *(const v16h*)(B2 + off);
        }

        // phase 2: 16 WMMAs (each B fragment consumed back-to-back)
        v8f c11 = {}, c12 = {}, c21 = {}, c22 = {};
        #pragma unroll
        for (int kb = 0; kb < 4; ++kb) {
            c11 = __builtin_amdgcn_wmma_f32_16x16x32_f16(false, a1[kb], false, b1v[kb], (short)0, c11, false, false);
            c21 = __builtin_amdgcn_wmma_f32_16x16x32_f16(false, a2[kb], false, b1v[kb], (short)0, c21, false, false);
            c12 = __builtin_amdgcn_wmma_f32_16x16x32_f16(false, a1[kb], false, b2v[kb], (short)0, c12, false, false);
            c22 = __builtin_amdgcn_wmma_f32_16x16x32_f16(false, a2[kb], false, b2v[kb], (short)0, c22, false, false);
        }

        // phase 3: exp + accumulate (branch-free; v_exp co-executes with WMMA)
        #pragma unroll
        for (int r = 0; r < 8; ++r) {
            s11[r] += __expf(c11[r] * INV_TAU);
            s12[r] += __expf(c12[r] * INV_TAU);
            s21[r] += __expf(c21[r] * INV_TAU);
            s22[r] += __expf(c22[r] * INV_TAU);
        }
    }

    // reduce row sums across the 16 lanes holding each row
    #pragma unroll
    for (int r = 0; r < 8; ++r) {
        #pragma unroll
        for (int m = 1; m < 16; m <<= 1) {
            s11[r] += __shfl_xor(s11[r], m, 32);
            s12[r] += __shfl_xor(s12[r], m, 32);
            s21[r] += __shfl_xor(s21[r], m, 32);
            s22[r] += __shfl_xor(s22[r], m, 32);
        }
    }

    // lanes 0..7 own rows 0..7 (r=lane); lanes 24..31 own rows 8..15
    const bool activeLo = (lane < 8);
    const bool activeHi = (lane >= 24);
    if (activeLo || activeHi) {
        const int r   = activeLo ? lane : (lane - 24);
        const int row = activeLo ? lane : (lane - 16);   // 0..15
        float S11 = 0.f, S12 = 0.f, S21 = 0.f, S22 = 0.f;
        #pragma unroll
        for (int rr = 0; rr < 8; ++rr) {
            if (rr == r) { S11 = s11[rr]; S12 = s12[rr]; S21 = s21[rr]; S22 = s22[rr]; }
        }
        float4 v; v.x = S11; v.y = S12; v.z = S21; v.w = S22;
        psums[((size_t)rowTile * SPLITS + chunk) * 16 + row] = v;
    }
}

// ---------------------------------------------------------------------------
// Kernel 3: per-row loss. One wave per rowTile (64 blocks x 8 waves).
// Sums the SPLITS chunk partials, computes diagonal dots directly from the
// f16 embeddings (same inputs as the WMMA diag, f32 accumulation), emits
// per-rowTile partial loss.
// ---------------------------------------------------------------------------
__global__ __launch_bounds__(256) void rowloss_kernel(
    const _Float16* __restrict__ h1, const _Float16* __restrict__ h2,
    const float4* __restrict__ psums, float* __restrict__ partials)
{
    const int lane    = threadIdx.x & 31;
    const int wave    = threadIdx.x >> 5;
    const int rowTile = blockIdx.x * 8 + wave;   // 0..511
    const int row0    = rowTile * 16;

    float myLoss = 0.f;
    if (lane < 16) {
        const int row = row0 + lane;
        float S11 = 0.f, S12 = 0.f, S21 = 0.f, S22 = 0.f;
        #pragma unroll
        for (int c = 0; c < SPLITS; ++c) {
            float4 v = psums[((size_t)rowTile * SPLITS + c) * 16 + lane];
            S11 += v.x; S12 += v.y; S21 += v.z; S22 += v.w;
        }
        // diagonal dots
        const _Float16* r1 = h1 + (size_t)row * DIMD;
        const _Float16* r2 = h2 + (size_t)row * DIMD;
        float d11 = 0.f, d22 = 0.f, d12 = 0.f;
        #pragma unroll 8
        for (int k = 0; k < DIMD; ++k) {
            const float f1 = (float)r1[k];
            const float f2 = (float)r2[k];
            d11 = fmaf(f1, f1, d11);
            d22 = fmaf(f2, f2, d22);
            d12 = fmaf(f1, f2, d12);
        }
        const float den1 = S11 + S12 - __expf(d11 * INV_TAU);
        const float den2 = S22 + S21 - __expf(d22 * INV_TAU);
        // l1 = log(den1) - d12/tau ; l2 = log(den2) - d12/tau ; 0.5*(l1+l2)
        myLoss = 0.5f * (__logf(den1) + __logf(den2)) - d12 * INV_TAU;
    }

    #pragma unroll
    for (int m = 1; m < 32; m <<= 1) myLoss += __shfl_xor(myLoss, m, 32);
    if (lane == 0) partials[rowTile] = myLoss;
}

// ---------------------------------------------------------------------------
// Kernel 4: mean over 512 tile-partials.
// ---------------------------------------------------------------------------
__global__ __launch_bounds__(256) void finalize_kernel(
    const float* __restrict__ partials, float* __restrict__ out)
{
    __shared__ float red[256];
    const int t = threadIdx.x;
    red[t] = partials[t] + partials[t + 256];
    __syncthreads();
    for (int s = 128; s > 0; s >>= 1) {
        if (t < s) red[t] += red[t + s];
        __syncthreads();
    }
    if (t == 0) out[0] = red[0] / (float)N_ROWS;
}

// ---------------------------------------------------------------------------
extern "C" void kernel_launch(void* const* d_in, const int* in_sizes, int n_in,
                              void* d_out, int out_size, void* d_ws, size_t ws_size,
                              hipStream_t stream) {
    const float* z1 = (const float*)d_in[0];
    const float* z2 = (const float*)d_in[1];
    const float* w1 = (const float*)d_in[2];
    const float* b1 = (const float*)d_in[3];
    const float* w2 = (const float*)d_in[4];
    const float* b2 = (const float*)d_in[5];

    _Float16* h1n = (_Float16*)d_ws;                         // 2 MB
    _Float16* h2n = h1n + (size_t)N_ROWS * DIMD;             // 2 MB
    float4* psums = (float4*)(h2n + (size_t)N_ROWS * DIMD);  // 512*16*16 float4 = 2 MB
    float* partials = (float*)(psums + (size_t)(NTILES) * SPLITS * 16);

    proj_norm_kernel<<<2 * N_ROWS, 128, 0, stream>>>(z1, z2, w1, b1, w2, b2, h1n, h2n);
    semi_loss_partial_kernel<<<64 * SPLITS, 256, 0, stream>>>(h1n, h2n, psums);
    rowloss_kernel<<<NTILES / 8, 256, 0, stream>>>(h1n, h2n, psums, partials);
    finalize_kernel<<<1, 256, 0, stream>>>(partials, (float*)d_out);
}
